// Transformer_14078902796835
// MI455X (gfx1250) — matrix operand order repaired and checked by reading, not compiled, not run on hardware
//
#include <hip/hip_runtime.h>
#include <hip/hip_bf16.h>
#include <cstdint>

// Problem constants (match reference)
#define NN   128
#define LL   512
#define NBLK 4
#define DD   1024
#define HH   16
#define DHH  64
#define TT   512
#define LCHUNK 8
#define GROWS  4          // rows per TDM tile

typedef __attribute__((ext_vector_type(16))) _Float16 v16h;
typedef __attribute__((ext_vector_type(8)))  _Float16 v8h;
typedef __attribute__((ext_vector_type(4)))  _Float16 v4h;
typedef __attribute__((ext_vector_type(8)))  float    v8f;
typedef unsigned int u32x4 __attribute__((ext_vector_type(4)));
typedef unsigned int u32x8 __attribute__((ext_vector_type(8)));

// ---------------------------------------------------------------------------
// Generic WMMA GEMM:  C[M,N] (+)= act( alpha * A[M,K] @ B(K,N) + bias )
// fp32 in global; f16 staged in LDS; f32 accumulate via v_wmma_f32_16x16x32_f16.
// Fast path (full tiles, no transB): b128 global loads + b64 LDS stores.
// ---------------------------------------------------------------------------
#define TM 64
#define TN 128
#define TK 32

__global__ __launch_bounds__(256) void gemm_wmma(
    const float* __restrict__ A, int lda, long strideA,
    const float* __restrict__ B, int ldb, long strideB,
    float*       __restrict__ C, int ldc, long strideC,
    const float* __restrict__ bias,
    int M, int Np, int K,
    int transB, int accum, int act, float alpha)
{
  __shared__ __align__(16) _Float16 tA[TM][TK];   // 4 KB
  __shared__ __align__(16) _Float16 tB[TN][TK];   // 8 KB, stored [n][k]
  int bz = blockIdx.z;
  A += (long)bz * strideA;
  B += (long)bz * strideB;
  C += (long)bz * strideC;
  int m0 = blockIdx.y * TM, n0 = blockIdx.x * TN;
  int tid  = threadIdx.x;
  int lane = tid & 31, wave = tid >> 5;
  int wm = (wave >> 2) * 32;        // 0 / 32
  int wn = (wave & 3) * 32;         // 0 / 32 / 64 / 96
  int lr = lane & 15, hf = lane >> 4;

  bool full = (!transB) && (m0 + TM <= M) && (n0 + TN <= Np) && ((K & (TK - 1)) == 0);

  v8f acc[2][2] = {};

  for (int k0 = 0; k0 < K; k0 += TK) {
    if (full) {
      // ---- A tile 64x32: thread -> rows r, r+32; 4 consecutive k ----
      int r  = tid >> 3;            // 0..31
      int kq = (tid & 7) * 4;       // 0..28
      float4 a0 = *(const float4*)&A[(long)(m0 + r) * lda + k0 + kq];
      float4 a1 = *(const float4*)&A[(long)(m0 + r + 32) * lda + k0 + kq];
      v4h ah0 = {(_Float16)a0.x, (_Float16)a0.y, (_Float16)a0.z, (_Float16)a0.w};
      v4h ah1 = {(_Float16)a1.x, (_Float16)a1.y, (_Float16)a1.z, (_Float16)a1.w};
      *(v4h*)&tA[r][kq]      = ah0;
      *(v4h*)&tA[r + 32][kq] = ah1;
      // ---- B tile 32x128 -> [n][k]: 4 n x 4 k micro-transpose ----
      int ng = (tid & 31) * 4;      // 0..124
      int kg = (tid >> 5) * 4;      // 0..28
      const float* Bp = &B[(long)(k0 + kg) * ldb + n0 + ng];
      float4 b0 = *(const float4*)(Bp);
      float4 b1 = *(const float4*)(Bp + ldb);
      float4 b2 = *(const float4*)(Bp + 2 * (long)ldb);
      float4 b3 = *(const float4*)(Bp + 3 * (long)ldb);
      v4h c0 = {(_Float16)b0.x, (_Float16)b1.x, (_Float16)b2.x, (_Float16)b3.x};
      v4h c1 = {(_Float16)b0.y, (_Float16)b1.y, (_Float16)b2.y, (_Float16)b3.y};
      v4h c2 = {(_Float16)b0.z, (_Float16)b1.z, (_Float16)b2.z, (_Float16)b3.z};
      v4h c3 = {(_Float16)b0.w, (_Float16)b1.w, (_Float16)b2.w, (_Float16)b3.w};
      *(v4h*)&tB[ng + 0][kg] = c0;
      *(v4h*)&tB[ng + 1][kg] = c1;
      *(v4h*)&tB[ng + 2][kg] = c2;
      *(v4h*)&tB[ng + 3][kg] = c3;
      if (k0 + TK < K) {
        __builtin_prefetch(&A[(long)(m0 + r) * lda + k0 + TK + kq], 0, 0);
        __builtin_prefetch(&B[(long)(k0 + TK + kg) * ldb + n0 + ng], 0, 0);
      }
    } else {
      for (int i = tid; i < TM * TK; i += 256) {
        int r = i >> 5, cc = i & (TK - 1);
        float v = 0.f;
        if (m0 + r < M && k0 + cc < K) v = A[(long)(m0 + r) * lda + (k0 + cc)];
        tA[r][cc] = (_Float16)v;
      }
      for (int i = tid; i < TN * TK; i += 256) {
        int nn = i >> 5, kk = i & (TK - 1);
        float v = 0.f;
        if (n0 + nn < Np && k0 + kk < K)
          v = transB ? B[(long)(n0 + nn) * ldb + (k0 + kk)]
                     : B[(long)(k0 + kk) * ldb + (n0 + nn)];
        tB[nn][kk] = (_Float16)v;
      }
    }
    __syncthreads();

#pragma unroll
    for (int mi = 0; mi < 2; ++mi) {
      // A fragment 16x32: row = lane%16; halves j<8 -> K=hf*8+j, j>=8 -> K=16+hf*8+j
      const v8h* pa0 = (const v8h*)&tA[wm + mi * 16 + lr][hf * 8];
      const v8h* pa1 = (const v8h*)&tA[wm + mi * 16 + lr][16 + hf * 8];
      v8h alo = *pa0, ahi = *pa1;
      v16h af;
#pragma unroll
      for (int j = 0; j < 8; ++j) { af[j] = alo[j]; af[j + 8] = ahi[j]; }
#pragma unroll
      for (int ni = 0; ni < 2; ++ni) {
        // B fragment 32x16: col = lane%16; halves j<8 -> K=hf*8+j, j>=8 -> K=16+hf*8+j
        const v8h* pb0 = (const v8h*)&tB[wn + ni * 16 + lr][hf * 8];
        const v8h* pb1 = (const v8h*)&tB[wn + ni * 16 + lr][16 + hf * 8];
        v8h blo = *pb0, bhi = *pb1;
        v16h bf;
#pragma unroll
        for (int j = 0; j < 8; ++j) { bf[j] = blo[j]; bf[j + 8] = bhi[j]; }
        acc[mi][ni] = __builtin_amdgcn_wmma_f32_16x16x32_f16(
            false, af, false, bf, (short)0, acc[mi][ni], false, false);
      }
    }
    __syncthreads();
  }

  // C/D layout: VGPR r -> M = base + hf*8 + r, N = base + lane%16
#pragma unroll
  for (int mi = 0; mi < 2; ++mi)
#pragma unroll
    for (int ni = 0; ni < 2; ++ni) {
      int cn = n0 + wn + ni * 16 + lr;
      if (cn >= Np) continue;
#pragma unroll
      for (int r = 0; r < 8; ++r) {
        int cm = m0 + wm + mi * 16 + hf * 8 + r;
        if (cm >= M) continue;
        float v = acc[mi][ni][r] * alpha;
        if (bias) v += bias[cn];
        float* cp = &C[(long)cm * ldc + cn];
        if (accum) v += *cp;
        if (act == 1) v = fmaxf(v, 0.f);
        *cp = v;
      }
    }
}

// ---------------------------------------------------------------------------
// Positional encoding table
// ---------------------------------------------------------------------------
__global__ void build_pos(float* __restrict__ pos)
{
  int t = blockIdx.x, j = threadIdx.x;                  // 512 threads
  float inv = __expf(-(float)j * 0.017988946f);         // ln(10000)/512
  float s = (float)(TT - 1 - t) * inv;
  pos[(long)t * DD + j]       = __sinf(s);
  pos[(long)t * DD + 512 + j] = __cosf(s);
}

// ---------------------------------------------------------------------------
// Row layernorm over cols
// ---------------------------------------------------------------------------
__global__ __launch_bounds__(256) void layernorm_k(
    const float* __restrict__ X, float* __restrict__ Y,
    const float* __restrict__ g, const float* __restrict__ b, int cols)
{
  __shared__ float red[256];
  int row = blockIdx.x, tid = threadIdx.x;
  const float* x = X + (long)row * cols;
  float s = 0.f;
  for (int c = tid; c < cols; c += 256) s += x[c];
  red[tid] = s; __syncthreads();
  for (int st = 128; st > 0; st >>= 1) { if (tid < st) red[tid] += red[tid + st]; __syncthreads(); }
  float mean = red[0] / cols; __syncthreads();
  float s2 = 0.f;
  for (int c = tid; c < cols; c += 256) { float v = x[c] - mean; s2 += v * v; }
  red[tid] = s2; __syncthreads();
  for (int st = 128; st > 0; st >>= 1) { if (tid < st) red[tid] += red[tid + st]; __syncthreads(); }
  float rstd = rsqrtf(red[0] / cols + 1e-5f);
  for (int c = tid; c < cols; c += 256)
    Y[(long)row * cols + c] = (x[c] - mean) * rstd * g[c] + b[c];
}

// ---------------------------------------------------------------------------
// TDM: issue a 2D tensor_load_to_lds of GROWS rows x DD f32, row stride NB*D.
// D# built in SGPRs per CDNA5 ISA (groups 0..3).
// ---------------------------------------------------------------------------
__device__ __forceinline__ void tdm_load_rows(const float* gsrc, unsigned lds_byte_off)
{
  unsigned long long ga = (unsigned long long)(uintptr_t)gsrc;
  u32x4 g0;
  g0[0] = 1u;                                   // count=1, user mode
  g0[1] = lds_byte_off;                         // lds_addr
  g0[2] = (unsigned)ga;                         // global_addr[31:0]
  g0[3] = ((unsigned)(ga >> 32) & 0x01FFFFFFu)  // global_addr[56:32]
          | 0x80000000u;                        // type=2 ("image") at [127:126]
  u32x8 g1;
  g1[0] = 0x20000u;                             // data_size=2 (4 bytes)
  g1[1] = ((unsigned)DD & 0xFFFFu) << 16;       // tensor_dim0 lo16 @ [63:48]
  g1[2] = ((unsigned)DD >> 16);                 // tensor_dim0 hi16 @ [79:64]
  g1[3] = 0x4000u                               // tensor_dim1 hi16 (dim1 = 1<<30)
          | (((unsigned)DD & 0xFFFFu) << 16);   // tile_dim0 @ [127:112]
  g1[4] = (unsigned)GROWS;                      // tile_dim1 @ [143:128]
  g1[5] = (unsigned)(NBLK * DD);                // tensor_dim0_stride lo32
  g1[6] = 0u;                                   // stride0 hi16 | stride1 lo16
  g1[7] = 0u;                                   // stride1 hi
  u32x4 z = {0u, 0u, 0u, 0u};                   // groups 2/3: dims 2..4 unused
  asm volatile("tensor_load_to_lds %0, %1, %2, %3"
               :: "s"(g0), "s"(g1), "s"(z), "s"(z) : "memory");
}

// ---------------------------------------------------------------------------
// Streaming attention (online softmax), one WG per (n, L-chunk).
// mem rows: TDM double-buffered into LDS (TENSORcnt).
// pos rows: per-lane async loads into LDS (ASYNCcnt).
// ---------------------------------------------------------------------------
__global__ __launch_bounds__(256) void attn_stream(
    const float* __restrict__ mem, const int* __restrict__ mask,
    const int* __restrict__ midx, const float* __restrict__ pos,
    const float* __restrict__ Qk,
    const float* __restrict__ lg, const float* __restrict__ lb,
    float* __restrict__ accP, float* __restrict__ mP, float* __restrict__ dP,
    int blk)
{
  __shared__ __align__(16) float buf[2 * GROWS * DD];   // 32 KB double buffer
  __shared__ __align__(16) float rawp[DD];              // 4 KB
  __shared__ float kv[DD];                              // 4 KB
  __shared__ float acc[DD];                             // 4 KB
  __shared__ float redA[8], redB[8];
  __shared__ float mrun[HH], drun[HH], corrS[HH], wS[HH];
  int n = blockIdx.x, c = blockIdx.y, tid = threadIdx.x;
  int lane = tid & 31, wv = tid >> 5;

  for (int i = tid; i < DD; i += 256) acc[i] = 0.f;
  if (tid < HH) { mrun[tid] = -3.0e38f; drun[tid] = 0.f; }
  float qv[4], gg[4], bb[4];
#pragma unroll
  for (int i = 0; i < 4; ++i) {
    qv[i] = Qk[(long)n * DD + tid * 4 + i];   // softmax scale folded in
    gg[i] = lg[tid * 4 + i];
    bb[i] = lb[tid * 4 + i];
  }
  unsigned ldp     = (unsigned)(uintptr_t)&rawp[tid * 4];
  unsigned bufbase = (unsigned)(uintptr_t)&buf[0];

  const int ROWS = LL / LCHUNK;   // 64 rows per WG
  const int NG   = ROWS / GROWS;  // 16 groups
  int l0 = c * ROWS;

  // preload group 0 (wave 0 issues; TDM ignores EXEC, one issue per wave)
  if (wv == 0) {
    tdm_load_rows(mem + ((long)((n * LL + l0) * NBLK + blk)) * DD, bufbase);
    asm volatile("s_wait_tensorcnt 0x0" ::: "memory");
  }
  __syncthreads();

  int cur = 0;
  for (int g = 0; g < NG; ++g) {
    int nxt = cur ^ (GROWS * DD);
    if (wv == 0 && g + 1 < NG) {
      int ln = l0 + (g + 1) * GROWS;
      tdm_load_rows(mem + ((long)((n * LL + ln) * NBLK + blk)) * DD,
                    bufbase + (unsigned)nxt * 4u);
    }
    for (int r = 0; r < GROWS; ++r) {
      int l = l0 + g * GROWS + r;
      const float* pr = pos + (long)midx[n * LL + l] * DD;
      asm volatile("global_load_async_to_lds_b128 %0, %1, off"
                   :: "v"(ldp), "v"((unsigned long long)(uintptr_t)(pr + tid * 4))
                   : "memory");
      asm volatile("s_wait_asynccnt 0x0" ::: "memory");

      float vals[4]; float s = 0.f, s2 = 0.f;
#pragma unroll
      for (int i = 0; i < 4; ++i) {
        float v = buf[cur + r * DD + tid * 4 + i] + rawp[tid * 4 + i];
        vals[i] = v; s += v; s2 += v * v;
      }
      // wave32 reduction, then cross-wave combine
#pragma unroll
      for (int off = 16; off > 0; off >>= 1) {
        s  += __shfl_down(s,  off);
        s2 += __shfl_down(s2, off);
      }
      if (lane == 0) { redA[wv] = s; redB[wv] = s2; }
      __syncthreads();
      float st = 0.f, st2 = 0.f;
#pragma unroll
      for (int j = 0; j < 8; ++j) { st += redA[j]; st2 += redB[j]; }
      float mean = st * (1.f / DD);
      float var  = st2 * (1.f / DD) - mean * mean;
      float rstd = rsqrtf(var + 1e-5f);

      // LN + per-head energy partial (thread owns d = 4t..4t+3, one head)
      float p = 0.f;
#pragma unroll
      for (int i = 0; i < 4; ++i) {
        float kvv = (vals[i] - mean) * rstd * gg[i] + bb[i];
        kv[tid * 4 + i] = kvv;
        p += qv[i] * kvv;
      }
      p += __shfl_down(p, 8, 16);
      p += __shfl_down(p, 4, 16);
      p += __shfl_down(p, 2, 16);
      p += __shfl_down(p, 1, 16);
      if ((tid & 15) == 0) {
        int h = tid >> 4;
        float e = p;
        if (mask[n * LL + l] == 0) e = -1.0e30f;
        float mo = mrun[h];
        float mn = fmaxf(mo, e);
        float co = __expf(mo - mn);
        float w  = __expf(e - mn);
        mrun[h] = mn;
        drun[h] = drun[h] * co + w;
        corrS[h] = co; wS[h] = w;
      }
      __syncthreads();
#pragma unroll
      for (int i = 0; i < 4; ++i) {
        int d = tid + i * 256; int hh = d >> 6;
        acc[d] = acc[d] * corrS[hh] + wS[hh] * kv[d];
      }
      __syncthreads();
    }
    if (wv == 0) asm volatile("s_wait_tensorcnt 0x0" ::: "memory");
    __syncthreads();
    cur = nxt;
  }
  long base = ((long)n * LCHUNK + c) * DD;
#pragma unroll
  for (int i = 0; i < 4; ++i) { int d = tid + i * 256; accP[base + d] = acc[d]; }
  if (tid < HH) {
    mP[(n * LCHUNK + c) * HH + tid] = mrun[tid];
    dP[(n * LCHUNK + c) * HH + tid] = drun[tid];
  }
}

__global__ __launch_bounds__(256) void attn_combine(
    const float* __restrict__ accP, const float* __restrict__ mP,
    const float* __restrict__ dP, float* __restrict__ attkv)
{
  __shared__ float wjS[LCHUNK][HH];
  __shared__ float denS[HH];
  int n = blockIdx.x, tid = threadIdx.x;
  if (tid < HH) {
    float M = -3.0e38f;
    for (int j = 0; j < LCHUNK; ++j) M = fmaxf(M, mP[(n * LCHUNK + j) * HH + tid]);
    float den = 0.f;
    for (int j = 0; j < LCHUNK; ++j) {
      float w = __expf(mP[(n * LCHUNK + j) * HH + tid] - M);
      wjS[j][tid] = w;
      den += w * dP[(n * LCHUNK + j) * HH + tid];
    }
    denS[tid] = den;
  }
  __syncthreads();
#pragma unroll
  for (int i = 0; i < 4; ++i) {
    int d = tid + i * 256; int hh = d >> 6;
    float num = 0.f;
    for (int j = 0; j < LCHUNK; ++j)
      num += wjS[j][hh] * accP[((long)n * LCHUNK + j) * DD + d];
    attkv[(long)n * DD + d] = num / denS[hh];
  }
}

// ---------------------------------------------------------------------------
// GRU elementwise pieces
// ---------------------------------------------------------------------------
__global__ void gru_gate_r(const float* __restrict__ rpre,
                           const float* __restrict__ x,
                           float* __restrict__ t, int ntot)
{
  int i = blockIdx.x * blockDim.x + threadIdx.x;
  if (i < ntot) {
    float r = 1.f / (1.f + __expf(-rpre[i]));
    t[i] = r * x[i];
  }
}

__global__ void gru_out_k(const float* __restrict__ zpre,
                          const float* __restrict__ gpre,
                          const float* __restrict__ x,
                          const float* __restrict__ bg,
                          float* __restrict__ out, int ntot)
{
  int i = blockIdx.x * blockDim.x + threadIdx.x;
  if (i < ntot) {
    float z = 1.f / (1.f + __expf(-(zpre[i] - bg[i & (DD - 1)])));
    float hc = tanhf(gpre[i]);
    out[i] = (1.f - z) * x[i] + z * hc;
  }
}

__global__ void copy_li(const float* __restrict__ h, float* __restrict__ outli, int blk)
{
  int n = blockIdx.x, tid = threadIdx.x;
#pragma unroll
  for (int i = 0; i < 4; ++i) {
    int d = tid + i * 256;
    outli[((long)n * NBLK + blk) * DD + d] = h[(long)n * DD + d];
  }
}

// ---------------------------------------------------------------------------
extern "C" void kernel_launch(void* const* d_in, const int* in_sizes, int n_in,
                              void* d_out, int out_size, void* d_ws, size_t ws_size,
                              hipStream_t stream)
{
  (void)in_sizes; (void)n_in; (void)out_size; (void)ws_size;
  const float* x        = (const float*)d_in[0];
  const float* memories = (const float*)d_in[1];
  const int*   mask     = (const int*)d_in[2];
  const int*   midx     = (const int*)d_in[3];
  const float* We   = (const float*)d_in[5];
  const float* be   = (const float*)d_in[6];
  const float* Wq   = (const float*)d_in[7];
  const float* Wk   = (const float*)d_in[8];
  const float* Wv   = (const float*)d_in[9];
  const float* Wo   = (const float*)d_in[10];
  const float* bo   = (const float*)d_in[11];
  const float* Wfc  = (const float*)d_in[12];
  const float* bfc  = (const float*)d_in[13];
  const float* ln1_g = (const float*)d_in[14];
  const float* ln1_b = (const float*)d_in[15];
  const float* ln2_g = (const float*)d_in[16];
  const float* ln2_b = (const float*)d_in[17];
  const float* lnkv_g = (const float*)d_in[18];
  const float* lnkv_b = (const float*)d_in[19];
  const float* g1w[7]; for (int j = 0; j < 7; ++j) g1w[j] = (const float*)d_in[20 + j];
  const float* g2w[7]; for (int j = 0; j < 7; ++j) g2w[j] = (const float*)d_in[27 + j];

  float* out = (float*)d_out;
  float* ws  = (float*)d_ws;

  // workspace layout (floats) ~16.6 MB total
  float* POS   = ws;                         // TT*DD
  float* HB    = POS + (long)TT * DD;        // N*D
  float* LNB   = HB   + (long)NN * DD;
  float* QKb   = LNB  + (long)NN * DD;
  float* ATTKV = QKb  + (long)NN * DD;
  float* AO    = ATTKV + (long)NN * DD;
  float* RPRE  = AO   + (long)NN * DD;
  float* ZPRE  = RPRE + (long)NN * DD;
  float* GPRE  = ZPRE + (long)NN * DD;
  float* TBUF  = GPRE + (long)NN * DD;
  float* H1    = TBUF + (long)NN * DD;
  float* FWD   = H1   + (long)NN * DD;
  float* WQK   = FWD  + (long)NN * DD;       // H*DH*DH
  float* WVO   = WQK  + (long)HH * DHH * DHH;// D*D
  float* ACCP  = WVO  + (long)DD * DD;       // N*LCHUNK*D
  float* MPb   = ACCP + (long)NN * LCHUNK * DD;
  float* DPb   = MPb  + (long)NN * LCHUNK * HH;

  dim3 blk256(256);
  const float scale = 0.03125f; // 1/sqrt(D), folded into Qk
  float* outli = out + (long)NN * DD;
  int nelem = NN * DD;

  build_pos<<<dim3(TT), dim3(512), 0, stream>>>(POS);

  // h = relu(x @ We + be)
  gemm_wmma<<<dim3(DD / TN, NN / TM, 1), blk256, 0, stream>>>(
      x, DD, 0, We, DD, 0, HB, DD, 0, be, NN, DD, DD, 0, 0, 1, 1.f);

  for (int i = 0; i < NBLK; ++i) {
    copy_li<<<dim3(NN), blk256, 0, stream>>>(HB, outli, i);

    // Wvo = blockdiag(Wv[i]) @ Wo[i]   (batched over heads)
    gemm_wmma<<<dim3(DD / TN, 1, HH), blk256, 0, stream>>>(
        Wv + (long)i * HH * DHH * DHH, DHH, (long)DHH * DHH,
        Wo + (long)i * DD * DD,        DD,  (long)DHH * DD,
        WVO,                           DD,  (long)DHH * DD,
        nullptr, DHH, DD, DHH, 0, 0, 0, 1.f);

    // Wqk[h] = Wq[i][h] @ Wk[i][h]^T
    gemm_wmma<<<dim3(1, 1, HH), blk256, 0, stream>>>(
        Wq + (long)i * HH * DHH * DHH, DHH, (long)DHH * DHH,
        Wk + (long)i * HH * DHH * DHH, DHH, (long)DHH * DHH,
        WQK,                           DHH, (long)DHH * DHH,
        nullptr, DHH, DHH, DHH, 1, 0, 0, 1.f);

    // q_ = LN(h; ln1)
    layernorm_k<<<dim3(NN), blk256, 0, stream>>>(HB, LNB,
        ln1_g + (long)i * DD, ln1_b + (long)i * DD, DD);

    // Qk = scale * q_ @ Wqk   (batched heads, block-diagonal)
    gemm_wmma<<<dim3(1, NN / TM, HH), blk256, 0, stream>>>(
        LNB, DD, (long)DHH,
        WQK, DHH, (long)DHH * DHH,
        QKb, DD, (long)DHH,
        nullptr, NN, DHH, DHH, 0, 0, 0, scale);

    // streaming online-softmax attention over memories
    attn_stream<<<dim3(NN, LCHUNK), blk256, 0, stream>>>(
        memories, mask, midx, POS, QKb,
        lnkv_g + (long)i * DD, lnkv_b + (long)i * DD,
        ACCP, MPb, DPb, i);
    attn_combine<<<dim3(NN), blk256, 0, stream>>>(ACCP, MPb, DPb, ATTKV);

    // ao = attkv @ Wvo + bo[i]
    gemm_wmma<<<dim3(DD / TN, NN / TM, 1), blk256, 0, stream>>>(
        ATTKV, DD, 0, WVO, DD, 0, AO, DD, 0, bo + (long)i * DD,
        NN, DD, DD, 0, 0, 0, 1.f);

    // ---- GRU1: x = h (HB), y = ao ----
    gemm_wmma<<<dim3(DD / TN, NN / TM, 1), blk256, 0, stream>>>(
        AO, DD, 0, g1w[0] + (long)i * DD * DD, DD, 0, RPRE, DD, 0,
        nullptr, NN, DD, DD, 0, 0, 0, 1.f);
    gemm_wmma<<<dim3(DD / TN, NN / TM, 1), blk256, 0, stream>>>(
        HB, DD, 0, g1w[1] + (long)i * DD * DD, DD, 0, RPRE, DD, 0,
        nullptr, NN, DD, DD, 0, 1, 0, 1.f);
    gemm_wmma<<<dim3(DD / TN, NN / TM, 1), blk256, 0, stream>>>(
        AO, DD, 0, g1w[2] + (long)i * DD * DD, DD, 0, ZPRE, DD, 0,
        nullptr, NN, DD, DD, 0, 0, 0, 1.f);
    gemm_wmma<<<dim3(DD / TN, NN / TM, 1), blk256, 0, stream>>>(
        HB, DD, 0, g1w[3] + (long)i * DD * DD, DD, 0, ZPRE, DD, 0,
        nullptr, NN, DD, DD, 0, 1, 0, 1.f);
    gemm_wmma<<<dim3(DD / TN, NN / TM, 1), blk256, 0, stream>>>(
        AO, DD, 0, g1w[4] + (long)i * DD * DD, DD, 0, GPRE, DD, 0,
        nullptr, NN, DD, DD, 0, 0, 0, 1.f);
    gru_gate_r<<<dim3(nelem / 256), blk256, 0, stream>>>(RPRE, HB, TBUF, nelem);
    gemm_wmma<<<dim3(DD / TN, NN / TM, 1), blk256, 0, stream>>>(
        TBUF, DD, 0, g1w[5] + (long)i * DD * DD, DD, 0, GPRE, DD, 0,
        nullptr, NN, DD, DD, 0, 1, 0, 1.f);
    gru_out_k<<<dim3(nelem / 256), blk256, 0, stream>>>(
        ZPRE, GPRE, HB, g1w[6] + (long)i * DD, H1, nelem);

    // h_ = LN(h1; ln2);  fwd = relu(h_ @ Wfc + bfc)
    layernorm_k<<<dim3(NN), blk256, 0, stream>>>(H1, LNB,
        ln2_g + (long)i * DD, ln2_b + (long)i * DD, DD);
    gemm_wmma<<<dim3(DD / TN, NN / TM, 1), blk256, 0, stream>>>(
        LNB, DD, 0, Wfc + (long)i * DD * DD, DD, 0, FWD, DD, 0,
        bfc + (long)i * DD, NN, DD, DD, 0, 0, 1, 1.f);

    // ---- GRU2: x = h1, y = fwd ----
    gemm_wmma<<<dim3(DD / TN, NN / TM, 1), blk256, 0, stream>>>(
        FWD, DD, 0, g2w[0] + (long)i * DD * DD, DD, 0, RPRE, DD, 0,
        nullptr, NN, DD, DD, 0, 0, 0, 1.f);
    gemm_wmma<<<dim3(DD / TN, NN / TM, 1), blk256, 0, stream>>>(
        H1, DD, 0, g2w[1] + (long)i * DD * DD, DD, 0, RPRE, DD, 0,
        nullptr, NN, DD, DD, 0, 1, 0, 1.f);
    gemm_wmma<<<dim3(DD / TN, NN / TM, 1), blk256, 0, stream>>>(
        FWD, DD, 0, g2w[2] + (long)i * DD * DD, DD, 0, ZPRE, DD, 0,
        nullptr, NN, DD, DD, 0, 0, 0, 1.f);
    gemm_wmma<<<dim3(DD / TN, NN / TM, 1), blk256, 0, stream>>>(
        H1, DD, 0, g2w[3] + (long)i * DD * DD, DD, 0, ZPRE, DD, 0,
        nullptr, NN, DD, DD, 0, 1, 0, 1.f);
    gemm_wmma<<<dim3(DD / TN, NN / TM, 1), blk256, 0, stream>>>(
        FWD, DD, 0, g2w[4] + (long)i * DD * DD, DD, 0, GPRE, DD, 0,
        nullptr, NN, DD, DD, 0, 0, 0, 1.f);
    gru_gate_r<<<dim3(nelem / 256), blk256, 0, stream>>>(RPRE, H1, TBUF, nelem);
    gemm_wmma<<<dim3(DD / TN, NN / TM, 1), blk256, 0, stream>>>(
        TBUF, DD, 0, g2w[5] + (long)i * DD * DD, DD, 0, GPRE, DD, 0,
        nullptr, NN, DD, DD, 0, 1, 0, 1.f);
    gru_out_k<<<dim3(nelem / 256), blk256, 0, stream>>>(
        ZPRE, GPRE, H1, g2w[6] + (long)i * DD, HB, nelem);
  }

  hipMemcpyAsync(out, HB, (size_t)NN * DD * sizeof(float),
                 hipMemcpyDeviceToDevice, stream);
}
